// MultiheadAttention_19129784336844
// MI455X (gfx1250) — matrix operand order repaired and checked by reading, not compiled, not run on hardware
//
#include <hip/hip_runtime.h>
#include <hip/hip_bf16.h>

// ---------------------------------------------------------------------------
// MHA forward for MI455X (gfx1250, wave32, WMMA).
// B=4, S=2048, D=1024, H=16, HD=64. All matrix math via v_wmma_f32_16x16x32_bf16.
// Round 3: async global->LDS DMA (ASYNCcnt path) for the two conversion-free
// staging paths (flash-attn K tile, out-proj A tile).
// ---------------------------------------------------------------------------

#define B_SZ   4
#define S_LEN  2048
#define D_MOD  1024
#define NH     16
#define HD     64

typedef __bf16 bf16_t;
typedef __attribute__((ext_vector_type(16))) __bf16 v16bf;
typedef __attribute__((ext_vector_type(8)))  float  v8f;

union FragBF { v16bf v; uint4 q[2]; };

// A-matrix 16x32 bf16 fragment (ISA 7.12.2): lane(m=lane&15, half=lane>>4):
//   elements 0..7  -> k = half*8 + e        (contiguous 16B)
//   elements 8..15 -> k = 16 + half*8 + e-8 (contiguous 16B)
__device__ __forceinline__ v16bf load_a_frag(const bf16_t* p, int stride, int lane) {
    FragBF f;
    int m = lane & 15, half = lane >> 4;
    const bf16_t* s = p + m * stride + half * 8;
    f.q[0] = *reinterpret_cast<const uint4*>(s);
    f.q[1] = *reinterpret_cast<const uint4*>(s + 16);
    return f.v;
}

// B-matrix 32x16 bf16 fragment from k-contiguous storage [n][k]:
//   lane(n=lane&15, half=lane>>4): elements 0..7 -> k = half*8 + e, 8..15 -> k = 16 + half*8 + e-8
__device__ __forceinline__ v16bf load_b_frag_kmajor(const bf16_t* p, int stride, int lane) {
    FragBF f;
    int n = lane & 15, half = lane >> 4;
    const bf16_t* s = p + n * stride + half * 8;
    f.q[0] = *reinterpret_cast<const uint4*>(s);
    f.q[1] = *reinterpret_cast<const uint4*>(s + 16);
    return f.v;
}

__device__ __forceinline__ v8f wmma_bf16(v16bf a, v16bf b, v8f c) {
    return __builtin_amdgcn_wmma_f32_16x16x32_bf16(false, a, false, b, (short)0, c,
                                                   false, false);
}

// Async global -> LDS 16B copy (GLOBAL_LOAD_ASYNC_TO_LDS_B128, ASYNCcnt).
// lds_off = workgroup-relative LDS byte offset (low 32 bits of generic addr).
__device__ __forceinline__ void async_copy_b128(unsigned lds_off, const void* gaddr) {
    asm volatile("global_load_async_to_lds_b128 %0, %1, off"
                 :: "v"(lds_off), "v"(gaddr)
                 : "memory");
}
__device__ __forceinline__ void wait_async0() {
    asm volatile("s_wait_asynccnt 0x0" ::: "memory");
}
__device__ __forceinline__ unsigned lds_offset(const void* p) {
    return (unsigned)(unsigned long long)p;   // generic LDS addr: offset in [31:0]
}

// ---------------------------------------------------------------------------
// Kernel 1: QKV projection. Y = x @ W  (M=8192, N=1024, K=1024)
// Block tile 128x256, 8 waves (2x4), wave tile 64x64. Q scaled by 1/sqrt(HD).
// W staged k-contiguous (sBt[n][k]) so B-frags are two ds_load_b128.
// Output layout [b][h][s][hd] bf16.
// ---------------------------------------------------------------------------
__global__ __launch_bounds__(256) void qkv_proj_kernel(
    const float* __restrict__ x, const float* __restrict__ Wq,
    const float* __restrict__ Wk, const float* __restrict__ Wv,
    bf16_t* __restrict__ qout, bf16_t* __restrict__ kout, bf16_t* __restrict__ vout)
{
    __shared__ __align__(16) bf16_t sA [128 * 40];   // [row][k]  128 x 32 (pad 40)
    __shared__ __align__(16) bf16_t sBt[256 * 40];   // [n][k]    256 x 32 (pad 40)

    const int tid  = threadIdx.x;
    const int lane = tid & 31;
    const int w    = tid >> 5;
    const int wm   = w >> 2;          // 0..1
    const int wn   = w & 3;           // 0..3

    const int which = blockIdx.y;
    const float* W  = (which == 0) ? Wq : (which == 1) ? Wk : Wv;
    bf16_t* dst     = (which == 0) ? qout : (which == 1) ? kout : vout;
    const float outscale = (which == 0) ? 0.125f : 1.0f;   // 1/sqrt(64) for Q

    const int m0 = (blockIdx.x >> 2) * 128;
    const int n0 = (blockIdx.x & 3) * 256;

    v8f acc[4][4];
#pragma unroll
    for (int i = 0; i < 4; ++i)
#pragma unroll
        for (int j = 0; j < 4; ++j) acc[i][j] = {};

    for (int kb = 0; kb < D_MOD; kb += 32) {
        {   // stage A (convert f32 -> bf16): 128 x 32, 16 floats per thread
            int row = tid >> 1, koff = (tid & 1) * 16;
            const float* src = x + (size_t)(m0 + row) * D_MOD + kb + koff;
            bf16_t* d = sA + row * 40 + koff;
#pragma unroll
            for (int i = 0; i < 16; ++i) d[i] = (bf16_t)src[i];
        }
        {   // stage B transposed: thread t owns column n0+t; per-k reads are
            // coalesced across threads (256 consecutive floats per k-row)
            const float* src = W + (size_t)kb * D_MOD + n0 + tid;
            union { bf16_t h[32]; uint4 q[4]; } buf;
#pragma unroll
            for (int k = 0; k < 32; ++k)
                buf.h[k] = (bf16_t)src[(size_t)k * D_MOD];
            uint4* d = reinterpret_cast<uint4*>(sBt + tid * 40);
#pragma unroll
            for (int i = 0; i < 4; ++i) d[i] = buf.q[i];
        }
        if (kb + 32 < D_MOD) {   // pull next k-block toward L2/L0
            __builtin_prefetch(x + (size_t)(m0 + (tid >> 1)) * D_MOD + kb + 32, 0, 1);
            __builtin_prefetch(W + (size_t)(kb + 32 + (tid & 31)) * D_MOD + n0 + (tid >> 5) * 32,
                               0, 1);
        }
        __syncthreads();

        v16bf a[4];
#pragma unroll
        for (int mc = 0; mc < 4; ++mc)
            a[mc] = load_a_frag(sA + (wm * 64 + mc * 16) * 40, 40, lane);
#pragma unroll
        for (int nc = 0; nc < 4; ++nc) {
            v16bf b = load_b_frag_kmajor(sBt + (wn * 64 + nc * 16) * 40, 40, lane);
#pragma unroll
            for (int mc = 0; mc < 4; ++mc)
                acc[mc][nc] = wmma_bf16(a[mc], b, acc[mc][nc]);
        }
        __syncthreads();
    }

    // Epilogue: D layout row = j + 8*half, col = lane&15. Write [b][h][s][hd].
    const int half = lane >> 4, nl = lane & 15;
#pragma unroll
    for (int mc = 0; mc < 4; ++mc)
#pragma unroll
        for (int nc = 0; nc < 4; ++nc)
#pragma unroll
            for (int j = 0; j < 8; ++j) {
                int m = m0 + wm * 64 + mc * 16 + j + 8 * half;
                int n = n0 + wn * 64 + nc * 16 + nl;
                int bb = m >> 11, ss = m & (S_LEN - 1);
                int hh = n >> 6,  dd = n & (HD - 1);
                dst[(((size_t)bb * NH + hh) * S_LEN + ss) * HD + dd] =
                    (bf16_t)(acc[mc][nc][j] * outscale);
            }
}

// ---------------------------------------------------------------------------
// Kernel 2: causal flash attention. One block per (b*H+h, 128 q rows).
// 8 waves; each wave owns 16 q rows with online softmax in fp32 registers.
// QK^T and P@V via WMMA; K tile staged via async DMA (no VGPR round trip);
// P bounced through per-wave LDS for C->A relayout.
// ---------------------------------------------------------------------------
__global__ __launch_bounds__(256) void flash_attn_kernel(
    const bf16_t* __restrict__ qs, const bf16_t* __restrict__ ks,
    const bf16_t* __restrict__ vs, bf16_t* __restrict__ ctx)
{
    __shared__ __align__(16) bf16_t sK [64 * 72];       // [kcol][hd]   (k-major for B frag)
    __shared__ __align__(16) bf16_t sVt[64 * 72];       // [hd][kcol]   (transposed)
    __shared__ __align__(16) bf16_t sP [8 * 16 * 72];   // per-wave P staging

    const int tid  = threadIdx.x;
    const int lane = tid & 31;
    const int w    = tid >> 5;
    const int half = lane >> 4, nl = lane & 15;

    const int bh   = blockIdx.x;              // 0..63
    const int qblk = blockIdx.y;              // 0..15
    const int bb   = bh >> 4, hh = bh & 15;

    const bf16_t* Qb = qs + (size_t)bh * S_LEN * HD;
    const bf16_t* Kb = ks + (size_t)bh * S_LEN * HD;
    const bf16_t* Vb = vs + (size_t)bh * S_LEN * HD;

    const int qbase = qblk * 128 + w * 16;

    v16bf qf[2];
    qf[0] = load_a_frag(Qb + (size_t)qbase * HD,      HD, lane);
    qf[1] = load_a_frag(Qb + (size_t)qbase * HD + 32, HD, lane);

    v8f   O[4];
    float mrow[8], lrow[8];
#pragma unroll
    for (int c = 0; c < 4; ++c) O[c] = {};
#pragma unroll
    for (int j = 0; j < 8; ++j) { mrow[j] = -1e30f; lrow[j] = 0.0f; }

    const int ktiles  = qblk * 2 + 2;               // block causal extent, 64-wide tiles
    const int my_last = (qbase + 15) >> 6;          // wave causal extent
    bf16_t* myP = sP + w * (16 * 72);
    const unsigned sK_base = lds_offset(sK);

    for (int kt = 0; kt < ktiles; ++kt) {
        if (kt + 1 < ktiles) {                       // prefetch next K/V tile
            __builtin_prefetch(Kb + (size_t)(kt + 1) * 64 * HD + tid * 16, 0, 1);
            __builtin_prefetch(Vb + (size_t)(kt + 1) * 64 * HD + tid * 16, 0, 1);
        }
        // ---- cooperative tile staging ----
        {
            const uint4* srcK = (const uint4*)(Kb + (size_t)kt * 64 * HD);
            const uint4* srcV = (const uint4*)(Vb + (size_t)kt * 64 * HD);
#pragma unroll
            for (int i = 0; i < 2; ++i) {
                int c = tid * 2 + i;                 // 512 uint4 chunks of 8 bf16
                int row = c >> 3, off = (c & 7) * 8;
                // K: straight copy -> async DMA directly into LDS
                async_copy_b128(sK_base + (unsigned)(row * 72 + off) * 2, srcK + c);
                // V: transposed in flight -> VGPR path
                uint4 vv = srcV[c];
                const bf16_t* pv = reinterpret_cast<const bf16_t*>(&vv);
#pragma unroll
                for (int e = 0; e < 8; ++e)          // transpose into [hd][kcol]
                    sVt[(off + e) * 72 + row] = pv[e];
            }
        }
        wait_async0();                               // K DMA landed in LDS
        __syncthreads();

        if (kt <= my_last) {
            // ---- scores = Q K^T (pre-scaled) ----
            v8f sc[4];
#pragma unroll
            for (int c = 0; c < 4; ++c) sc[c] = {};
#pragma unroll
            for (int kstep = 0; kstep < 2; ++kstep)
#pragma unroll
                for (int c = 0; c < 4; ++c) {
                    v16bf bk = load_b_frag_kmajor(sK + (c * 16) * 72 + kstep * 32, 72, lane);
                    sc[c] = wmma_bf16(qf[kstep], bk, sc[c]);
                }

            // ---- causal mask ----
            const int colbase = kt * 64;
#pragma unroll
            for (int c = 0; c < 4; ++c) {
                int col = colbase + c * 16 + nl;
#pragma unroll
                for (int j = 0; j < 8; ++j) {
                    int row = qbase + j + 8 * half;
                    if (col > row) sc[c][j] = -1e30f;
                }
            }

            // ---- online softmax: row max over 16 lanes of this half ----
            float mnew[8];
#pragma unroll
            for (int j = 0; j < 8; ++j) {
                float v = fmaxf(fmaxf(sc[0][j], sc[1][j]), fmaxf(sc[2][j], sc[3][j]));
                v = fmaxf(v, __shfl_xor(v, 1));
                v = fmaxf(v, __shfl_xor(v, 2));
                v = fmaxf(v, __shfl_xor(v, 4));
                v = fmaxf(v, __shfl_xor(v, 8));
                mnew[j] = fmaxf(mrow[j], v);
            }
            float rsum[8];
#pragma unroll
            for (int j = 0; j < 8; ++j) {
                float f = __expf(mrow[j] - mnew[j]);
                lrow[j] *= f;
#pragma unroll
                for (int c = 0; c < 4; ++c) O[c][j] *= f;
                mrow[j] = mnew[j];
                rsum[j] = 0.0f;
            }
            // ---- P = exp(s - m); stage to LDS in row-major for A-frag reload ----
#pragma unroll
            for (int c = 0; c < 4; ++c)
#pragma unroll
                for (int j = 0; j < 8; ++j) {
                    float p = __expf(sc[c][j] - mrow[j]);
                    rsum[j] += p;
                    myP[(j + 8 * half) * 72 + c * 16 + nl] = (bf16_t)p;
                }
#pragma unroll
            for (int j = 0; j < 8; ++j) {
                float v = rsum[j];
                v += __shfl_xor(v, 1);
                v += __shfl_xor(v, 2);
                v += __shfl_xor(v, 4);
                v += __shfl_xor(v, 8);
                lrow[j] += v;
            }
            // wave-private LDS RAW: wait for ds stores before A-frag reload
            asm volatile("s_wait_dscnt 0x0" ::: "memory");

            // ---- O += P @ V ----
#pragma unroll
            for (int kstep = 0; kstep < 2; ++kstep) {
                v16bf ap = load_a_frag(myP + kstep * 32, 72, lane);
#pragma unroll
                for (int c = 0; c < 4; ++c) {
                    v16bf bv = load_b_frag_kmajor(sVt + (c * 16) * 72 + kstep * 32, 72, lane);
                    O[c] = wmma_bf16(ap, bv, O[c]);
                }
            }
        }
        __syncthreads();
    }

    // ---- normalize and write ctx [b*S+s][h*HD+hd] bf16 ----
#pragma unroll
    for (int j = 0; j < 8; ++j) {
        float inv = 1.0f / lrow[j];
        int row = qbase + j + 8 * half;
        size_t base = ((size_t)bb * S_LEN + row) * D_MOD + hh * HD;
#pragma unroll
        for (int c = 0; c < 4; ++c)
            ctx[base + c * 16 + nl] = (bf16_t)(O[c][j] * inv);
    }
}

// ---------------------------------------------------------------------------
// Kernel 3: output projection. out = ctx @ Wo + bo   (fp32 out)
// A tile (already bf16) staged via async DMA; Wo transposed/converted in VALU.
// ---------------------------------------------------------------------------
__global__ __launch_bounds__(256) void out_proj_kernel(
    const bf16_t* __restrict__ ctx, const float* __restrict__ Wo,
    const float* __restrict__ bo, float* __restrict__ out)
{
    __shared__ __align__(16) bf16_t sA [128 * 40];
    __shared__ __align__(16) bf16_t sBt[256 * 40];   // [n][k] k-contiguous

    const int tid  = threadIdx.x;
    const int lane = tid & 31;
    const int w    = tid >> 5;
    const int wm   = w >> 2, wn = w & 3;

    const int m0 = (blockIdx.x >> 2) * 128;
    const int n0 = (blockIdx.x & 3) * 256;
    const unsigned sA_base = lds_offset(sA);

    v8f acc[4][4];
#pragma unroll
    for (int i = 0; i < 4; ++i)
#pragma unroll
        for (int j = 0; j < 4; ++j) acc[i][j] = {};

    for (int kb = 0; kb < D_MOD; kb += 32) {
        {   // stage A: bf16 copy -> async DMA straight into LDS
            int row = tid >> 1, koff = (tid & 1) * 16;
            const bf16_t* src = ctx + (size_t)(m0 + row) * D_MOD + kb + koff;
            unsigned dst = sA_base + (unsigned)(row * 40 + koff) * 2;
            async_copy_b128(dst,      src);
            async_copy_b128(dst + 16, src + 8);
        }
        {   // stage B transposed: thread t owns column n0+t (coalesced per k-row)
            const float* src = Wo + (size_t)kb * D_MOD + n0 + tid;
            union { bf16_t h[32]; uint4 q[4]; } buf;
#pragma unroll
            for (int k = 0; k < 32; ++k)
                buf.h[k] = (bf16_t)src[(size_t)k * D_MOD];
            uint4* d = reinterpret_cast<uint4*>(sBt + tid * 40);
#pragma unroll
            for (int i = 0; i < 4; ++i) d[i] = buf.q[i];
        }
        if (kb + 32 < D_MOD) {
            __builtin_prefetch(ctx + (size_t)(m0 + (tid >> 1)) * D_MOD + kb + 32, 0, 1);
            __builtin_prefetch(Wo + (size_t)(kb + 32 + (tid & 31)) * D_MOD + n0 + (tid >> 5) * 32,
                               0, 1);
        }
        wait_async0();
        __syncthreads();

        v16bf a[4];
#pragma unroll
        for (int mc = 0; mc < 4; ++mc)
            a[mc] = load_a_frag(sA + (wm * 64 + mc * 16) * 40, 40, lane);
#pragma unroll
        for (int nc = 0; nc < 4; ++nc) {
            v16bf b = load_b_frag_kmajor(sBt + (wn * 64 + nc * 16) * 40, 40, lane);
#pragma unroll
            for (int mc = 0; mc < 4; ++mc)
                acc[mc][nc] = wmma_bf16(a[mc], b, acc[mc][nc]);
        }
        __syncthreads();
    }

    const int half = lane >> 4, nl = lane & 15;
#pragma unroll
    for (int mc = 0; mc < 4; ++mc)
#pragma unroll
        for (int nc = 0; nc < 4; ++nc) {
            int n = n0 + wn * 64 + nc * 16 + nl;
            float bias = bo[n];
#pragma unroll
            for (int j = 0; j < 8; ++j) {
                int m = m0 + wm * 64 + mc * 16 + j + 8 * half;
                out[(size_t)m * D_MOD + n] = acc[mc][nc][j] + bias;
            }
        }
}

// ---------------------------------------------------------------------------
extern "C" void kernel_launch(void* const* d_in, const int* in_sizes, int n_in,
                              void* d_out, int out_size, void* d_ws, size_t ws_size,
                              hipStream_t stream) {
    const float* x  = (const float*)d_in[0];
    const float* Wq = (const float*)d_in[1];
    const float* Wk = (const float*)d_in[2];
    const float* Wv = (const float*)d_in[3];
    const float* Wo = (const float*)d_in[4];
    const float* bo = (const float*)d_in[5];
    float* out = (float*)d_out;

    const size_t qkv_elems = (size_t)B_SZ * NH * S_LEN * HD;   // 8.4M bf16 each
    bf16_t* qs  = (bf16_t*)d_ws;
    bf16_t* ks  = qs + qkv_elems;
    bf16_t* vs  = ks + qkv_elems;
    bf16_t* ctx = vs + qkv_elems;                              // [B*S][D] bf16

    // 1) QKV projections: grid = 64 m-tiles * 4 n-tiles, y over {Q,K,V}
    qkv_proj_kernel<<<dim3(256, 3), 256, 0, stream>>>(x, Wq, Wk, Wv, qs, ks, vs);

    // 2) causal flash attention: grid = (B*H, S/128)
    flash_attn_kernel<<<dim3(64, 16), 256, 0, stream>>>(qs, ks, vs, ctx);

    // 3) output projection + bias
    out_proj_kernel<<<dim3(256), 256, 0, stream>>>(ctx, Wo, bo, out);
}
